// TemporalRenderCrossIllu_53068615910055
// MI455X (gfx1250) — matrix operand order repaired and checked by reading, not compiled, not run on hardware
//
#include <hip/hip_runtime.h>
#include <hip/hip_fp16.h>

// ---------------------------------------------------------------------------
// CDNA5 / gfx1250 fused Swin-block: LN -> window attention (block-causal,
// flash-style) -> out-proj -> residual -> LN -> MLP(GELU) -> residual.
// Matrix math: v_wmma_f32_16x16x32_f16 (wave32).
// Data staging: global_load_async_to_lds_b128 (+ s_wait_asynccnt), so shared
// B/K/V tiles are fetched once per workgroup instead of once per wave.
// ---------------------------------------------------------------------------

typedef _Float16 h8  __attribute__((ext_vector_type(8)));
typedef _Float16 h16 __attribute__((ext_vector_type(16)));
typedef float    f8  __attribute__((ext_vector_type(8)));

#define NT   147456   // total tokens = 2*8*96*96
#define CDIM 192
#define HMLP 768

// -------------------- async copy: 16B global -> LDS ------------------------
// ISA §15.18.3 op 98: GLOBAL_LOAD_ASYNC_TO_LDS_B128 (ASYNCcnt). Per-lane LDS
// dest address in VDST, 64-bit global address in VADDR (GV mode). The generic
// pointer's low 32 bits are the wave-relative LDS byte address.
__device__ __forceinline__ void async_copy16(_Float16* lds, const _Float16* g) {
  const unsigned int      l  = (unsigned int)(unsigned long long)(uintptr_t)lds;
  const unsigned long long ga = (unsigned long long)(uintptr_t)g;
  asm volatile("global_load_async_to_lds_b128 %0, %1, off"
               :: "v"(l), "v"(ga) : "memory");
}
__device__ __forceinline__ void async_wait0() {
  asm volatile("s_wait_asynccnt 0x0" ::: "memory");
}

// -------------------- WMMA helpers (ISA 7.12.2 layouts) --------------------

// A-matrix 16x32 f16 layout: lane<16 holds row m=lane, K {0..7} in v0..3 and
// K {16..23} in v4..7; lane>=16 holds row m=lane-16, K {8..15} and {24..31}.
__device__ __forceinline__ h16 ldA(const _Float16* t, int ld) {
  const int lane = threadIdx.x & 31;
  const int m  = lane & 15;
  const int ko = (lane & 16) ? 8 : 0;
  const _Float16* p = t + m * ld + ko;
  h16 r;
  *(h8*)&r       = *(const h8*)p;         // K ko..ko+7
  *((h8*)&r + 1) = *(const h8*)(p + 16);  // K ko+16..ko+23
  return r;
}

// B-matrix 32x16 f16 layout (from B^T stored row-major NxK): lane<16 holds
// col n=lane with K {0..7} and {16..23}; lane>=16 holds col n=lane-16, K {8..15} and {24..31}.
__device__ __forceinline__ h16 ldB(const _Float16* t, int ld) {
  const int lane = threadIdx.x & 31;
  const int n  = lane & 15;
  const int ko = (lane & 16) ? 8 : 0;
  const _Float16* p = t + n * ld + ko;
  h16 r;
  *(h8*)&r       = *(const h8*)p;         // K ko..ko+7
  *((h8*)&r + 1) = *(const h8*)(p + 16);  // K ko+16..ko+23
  return r;
}

__device__ __forceinline__ f8 wmma16(h16 a, h16 b, f8 c) {
  return __builtin_amdgcn_wmma_f32_16x16x32_f16(false, a, false, b,
                                                (short)0, c, false, false);
}

__device__ __forceinline__ f8 zero8() {
  f8 z = {0.f, 0.f, 0.f, 0.f, 0.f, 0.f, 0.f, 0.f};
  return z;
}

// -------------------- LayerNorm (+ optional window permute) ----------------
// One wave per token (192 channels, 6 per lane), 8 waves per block.
__global__ void __launch_bounds__(256)
ln_kernel(const float* __restrict__ x, const float* __restrict__ g,
          const float* __restrict__ b, _Float16* __restrict__ out, int permute) {
  const int wave = threadIdx.x >> 5;
  const int lane = threadIdx.x & 31;
  const size_t tok = (size_t)blockIdx.x * 8 + wave;
  const float* xp = x + tok * CDIM;

  float v[6];
  float s = 0.f;
#pragma unroll
  for (int i = 0; i < 6; ++i) { v[i] = xp[lane + 32 * i]; s += v[i]; }
#pragma unroll
  for (int o = 16; o > 0; o >>= 1) s += __shfl_xor(s, o, 32);
  const float mu = s * (1.0f / 192.0f);

  float q = 0.f;
#pragma unroll
  for (int i = 0; i < 6; ++i) { float d = v[i] - mu; q += d * d; }
#pragma unroll
  for (int o = 16; o > 0; o >>= 1) q += __shfl_xor(q, o, 32);
  const float rstd = rsqrtf(q * (1.0f / 192.0f) + 1e-5f);

  size_t row = tok;
  if (permute) {                      // (b,d,h,w) -> (bp = b*144+hy*12+wx, l)
    int t = (int)tok;
    const int w = t % 96; t /= 96;
    const int h = t % 96; t /= 96;
    const int d = t % 8;  t /= 8;     // t is now b
    const int bp = t * 144 + (h >> 3) * 12 + (w >> 3);
    const int l  = d * 64 + (h & 7) * 8 + (w & 7);
    row = (size_t)bp * 512 + l;
  }
  _Float16* op = out + row * CDIM;
#pragma unroll
  for (int i = 0; i < 6; ++i) {
    const int c = lane + 32 * i;
    op[c] = (_Float16)((v[i] - mu) * rstd * g[c] + b[c]);
  }
}

// -------------------- fp32 -> f16 weight convert ---------------------------
__global__ void __launch_bounds__(256)
cvt_kernel(const float* __restrict__ in, _Float16* __restrict__ out, int n) {
  const int i = blockIdx.x * 256 + threadIdx.x;
  if (i < n) out[i] = (_Float16)in[i];
}

// -------------------- Generic WMMA GEMM with fused epilogues ---------------
// C[m,n] = sum_k A[m,k] * W[n,k] + bias[n]; A f16 (M,K) rm, W f16 (N,K) rm.
// Block = 4 waves; wave computes a 16(M) x 64(N) strip; block tile 64x64.
// K must be a multiple of 192. B is staged per 64x192 panel in LDS via async
// loads (shared by all 4 waves); A rows are wave-private -> direct global.
struct GemmOut {
  _Float16* q;  _Float16* k;  _Float16* vt;     // mode 0 (QKV split, V^T)
  const float* x; float* y;                     // mode 1 (out-proj + residual)
  _Float16* oh;                                 // mode 2 (fc1 + GELU)
  const float* resid; float* of;                // mode 3 (fc2 + residual)
};

template <int MODE>
__global__ void __launch_bounds__(128)
gemm_kernel(const _Float16* __restrict__ A, const _Float16* __restrict__ W,
            const float* __restrict__ bias, int N, int K, GemmOut io) {
  const int tid  = threadIdx.x;
  const int wave = tid >> 5;
  const int lane = tid & 31;
  const int m0 = blockIdx.x * 64 + wave * 16;
  const int n0 = blockIdx.y * 64;
  const _Float16* Ab = A + (size_t)m0 * K;

  __shared__ __align__(16) _Float16 bpan[64 * 192];   // 24 KB B panel

  f8 acc[4];
#pragma unroll
  for (int nt = 0; nt < 4; ++nt) acc[nt] = zero8();

  const int npan = K / 192;
  for (int p = 0; p < npan; ++p) {
    // cooperative async stage: 64 rows x 192 K f16 = 1536 x 16B chunks
#pragma unroll
    for (int i = 0; i < 12; ++i) {
      const int e   = (i * 128 + tid) * 8;      // f16 element index in panel
      const int row = e / 192, kk = e % 192;
      async_copy16(&bpan[e], W + (size_t)(n0 + row) * K + p * 192 + kk);
    }
    async_wait0();
    __syncthreads();
#pragma unroll
    for (int cc = 0; cc < 6; ++cc) {
      if (cc + 1 < 6) __builtin_prefetch(Ab + (size_t)(p * 6 + cc + 1) * 32, 0, 1);
      const h16 a = ldA(Ab + (size_t)(p * 6 + cc) * 32, K);
#pragma unroll
      for (int nt = 0; nt < 4; ++nt)
        acc[nt] = wmma16(a, ldB(&bpan[(nt * 16) * 192 + cc * 32], 192), acc[nt]);
    }
    __syncthreads();
  }

  // D layout: vgpr r, lane<16 -> (M=r, N=lane); lane>=16 -> (M=r+8, N=lane-16)
  const int col = lane & 15;
  const int rb  = (lane & 16) ? 8 : 0;
#pragma unroll
  for (int nt = 0; nt < 4; ++nt) {
    const int n  = n0 + nt * 16 + col;
    const float bv = bias[n];
    // mode-0 column decode hoisted out of the row loop
    int which = 0, head = 0, e = 0;
    if (MODE == 0) {
      which = n / 192;
      const int ch = n % 192;
      head = ch / 96;
      e = ch % 96;
    }
#pragma unroll
    for (int r = 0; r < 8; ++r) {
      const int m = m0 + rb + r;
      const float v = acc[nt][r] + bv;
      if (MODE == 0) {
        const int bp = m >> 9, l = m & 511;
        if (which == 2) {
          io.vt[((size_t)(bp * 2 + head) * 96 + e) * 512 + l] = (_Float16)v;
        } else {
          const size_t idx = ((size_t)(bp * 2 + head) * 512 + l) * 96 + e;
          if (which == 0) io.q[idx] = (_Float16)v;
          else            io.k[idx] = (_Float16)v;
        }
      } else if (MODE == 1) {
        // reverse window partition: (bp,l) -> (b,d,h,w) token, add residual
        const int bp = m >> 9, l = m & 511;
        const int b = bp / 144, p2 = bp % 144;
        const int hy = p2 / 12, wx = p2 % 12;
        const int d = l >> 6, rr = l & 63;
        const int hh = hy * 8 + (rr >> 3), ww = wx * 8 + (rr & 7);
        const size_t tok = ((size_t)(b * 8 + d) * 96 + hh) * 96 + ww;
        const size_t idx = tok * CDIM + n;
        io.y[idx] = v + io.x[idx];
      } else if (MODE == 2) {
        // exact GELU: 0.5*v*(1+erf(v/sqrt(2)))
        io.oh[(size_t)m * N + n] =
            (_Float16)(0.5f * v * (1.0f + erff(v * 0.70710678118f)));
      } else {
        const size_t idx = (size_t)m * N + n;
        io.of[idx] = v + io.resid[idx];
      }
    }
  }
}

// -------------------- Flash attention, block-causal ------------------------
// One block per (window*head, temporal q-block of 64 rows); 4 waves x 16 rows.
// Mask is structural: q-block t only iterates k-blocks 0..t. K tile (64x96)
// and V chunk (96x64) are staged in LDS once per k-step via async loads and
// shared by all 4 waves.
__global__ void __launch_bounds__(128)
attn_kernel(const _Float16* __restrict__ q, const _Float16* __restrict__ k,
            const _Float16* __restrict__ vt, _Float16* __restrict__ ao) {
  const int tblk = blockIdx.x & 7;
  const int prob = blockIdx.x >> 3;      // bp*2 + head
  const int head = prob & 1;
  const int bp   = prob >> 1;
  const int tid  = threadIdx.x;
  const int wave = tid >> 5;
  const int lane = tid & 31;

  __shared__ __align__(16) _Float16 lp[4][16 * 64];   // per-wave P tile  (8 KB)
  __shared__ __align__(16) _Float16 kbuf[64 * 96];    // K tile           (12 KB)
  __shared__ __align__(16) _Float16 vbuf[96 * 64];    // V^T chunk        (12 KB)

  const _Float16* qb = q  + (size_t)prob * 512 * 96 +
                       (size_t)(tblk * 64 + wave * 16) * 96;
  const _Float16* kb = k  + (size_t)prob * 512 * 96;
  const _Float16* vb = vt + (size_t)prob * 96 * 512;

  h16 aq[3];
#pragma unroll
  for (int c = 0; c < 3; ++c) aq[c] = ldA(qb + c * 32, 96);

  f8 o[6];
#pragma unroll
  for (int j = 0; j < 6; ++j) o[j] = zero8();

  // per-lane stats: index r means row (rb + r) of this wave's 16-row strip,
  // consistent with the WMMA C layout (lane-half selects rb = 0 or 8).
  float mrun[8], lrun[8];
#pragma unroll
  for (int r = 0; r < 8; ++r) { mrun[r] = -1e30f; lrun[r] = 0.f; }

  const float scale = 0.1020620726f;  // 1/sqrt(96)
  _Float16* lpw = &lp[wave][0];
  const int col = lane & 15;
  const int rb  = (lane & 16) ? 8 : 0;

  for (int kt = 0; kt <= tblk; ++kt) {
    // cooperative async stage of K tile (64x96) and V chunk (96x64):
    // 768 x 16B chunks each, 128 threads -> 6 issues apiece.
#pragma unroll
    for (int i = 0; i < 6; ++i) {
      const int e = (i * 128 + tid) * 8;              // f16 element index
      {
        const int row = e / 96, kk = e % 96;
        async_copy16(&kbuf[e], kb + (size_t)(kt * 64 + row) * 96 + kk);
      }
      {
        const int row = e >> 6, kk = e & 63;
        async_copy16(&vbuf[e], vb + (size_t)row * 512 + kt * 64 + kk);
      }
    }
    async_wait0();
    __syncthreads();

    f8 s[4];
#pragma unroll
    for (int nt = 0; nt < 4; ++nt) s[nt] = zero8();
#pragma unroll
    for (int nt = 0; nt < 4; ++nt) {
#pragma unroll
      for (int c = 0; c < 3; ++c)
        s[nt] = wmma16(aq[c], ldB(&kbuf[(nt * 16) * 96 + c * 32], 96), s[nt]);
    }
#pragma unroll
    for (int nt = 0; nt < 4; ++nt)
#pragma unroll
      for (int r = 0; r < 8; ++r) s[nt][r] *= scale;

    // online softmax, stats replicated per 16-lane half via xor-shuffles
#pragma unroll
    for (int r = 0; r < 8; ++r) {
      float mx = fmaxf(fmaxf(s[0][r], s[1][r]), fmaxf(s[2][r], s[3][r]));
#pragma unroll
      for (int off = 1; off < 16; off <<= 1) mx = fmaxf(mx, __shfl_xor(mx, off, 32));
      const float mn = fmaxf(mrun[r], mx);
      const float sc = __expf(mrun[r] - mn);
      float rs = 0.f;
#pragma unroll
      for (int nt = 0; nt < 4; ++nt) {
        const float pv = __expf(s[nt][r] - mn);
        s[nt][r] = pv;
        rs += pv;
      }
#pragma unroll
      for (int off = 1; off < 16; off <<= 1) rs += __shfl_xor(rs, off, 32);
      lrun[r] = lrun[r] * sc + rs;
      mrun[r] = mn;
#pragma unroll
      for (int j = 0; j < 6; ++j) o[j][r] *= sc;
    }

    // repack P (C layout) -> LDS row-major 16x64 for use as WMMA A
#pragma unroll
    for (int nt = 0; nt < 4; ++nt)
#pragma unroll
      for (int r = 0; r < 8; ++r)
        lpw[(rb + r) * 64 + nt * 16 + col] = (_Float16)s[nt][r];

    // O += P @ V  (B fed from staged V^T rows, K = 64 keys in 2 chunks);
    // lp is wave-private and LDS ops are in-order per wave, so no barrier
    // is needed between the P stores and these loads.
#pragma unroll
    for (int c2 = 0; c2 < 2; ++c2) {
      const h16 pa = ldA(lpw + c2 * 32, 64);
#pragma unroll
      for (int j = 0; j < 6; ++j) {
        const h16 bv = ldB(&vbuf[(j * 16) * 64 + c2 * 32], 64);
        o[j] = wmma16(pa, bv, o[j]);
      }
    }
    __syncthreads();   // protect kbuf/vbuf before next k-step overwrites
  }

  // normalize and write (bp, l, head*96 + ch) f16
#pragma unroll
  for (int r = 0; r < 8; ++r) {
    const float inv = 1.0f / lrun[r];
    const int l = tblk * 64 + wave * 16 + rb + r;
    _Float16* op = ao + ((size_t)bp * 512 + l) * CDIM + head * 96;
#pragma unroll
    for (int j = 0; j < 6; ++j) op[j * 16 + col] = (_Float16)(o[j][r] * inv);
  }
}

// -------------------- launcher --------------------------------------------
extern "C" void kernel_launch(void* const* d_in, const int* in_sizes, int n_in,
                              void* d_out, int out_size, void* d_ws, size_t ws_size,
                              hipStream_t stream) {
  (void)in_sizes; (void)n_in; (void)out_size; (void)ws_size;
  const float* x    = (const float*)d_in[0];
  const float* n3g  = (const float*)d_in[1];
  const float* n3b  = (const float*)d_in[2];
  const float* Win  = (const float*)d_in[3];   // (576,192)
  const float* bin  = (const float*)d_in[4];
  const float* Wout = (const float*)d_in[5];   // (192,192)
  const float* bout = (const float*)d_in[6];
  const float* n4g  = (const float*)d_in[7];
  const float* n4b  = (const float*)d_in[8];
  const float* Wf1  = (const float*)d_in[9];   // (768,192)
  const float* bf1  = (const float*)d_in[10];
  const float* Wf2  = (const float*)d_in[11];  // (192,768)
  const float* bf2  = (const float*)d_in[12];
  float* out = (float*)d_out;

  char* ws = (char*)d_ws;
  size_t off = 0;
  auto alloc = [&](size_t bytes) -> void* {
    void* p = ws + off;
    off += (bytes + 255) & ~(size_t)255;
    return p;
  };
  _Float16* r1 = (_Float16*)alloc((size_t)NT * CDIM * 2);  // h / attn_out / h2
  _Float16* r2 = (_Float16*)alloc((size_t)NT * HMLP * 2);  // qkv, then hidden
  float*    y  = (float*)alloc((size_t)NT * CDIM * 4);     // post-attn residual
  _Float16* w16 = (_Float16*)alloc((size_t)442368 * 2);    // all weights f16

  _Float16* win16  = w16;
  _Float16* wout16 = win16 + 110592;
  _Float16* wf1_16 = wout16 + 36864;
  _Float16* wf2_16 = wf1_16 + 147456;

  _Float16* qb  = r2;                             // (576, 512, 96)
  _Float16* kbf = qb  + (size_t)576 * 512 * 96;   // (576, 512, 96)
  _Float16* vtb = kbf + (size_t)576 * 512 * 96;   // (576, 96, 512)  V^T

  cvt_kernel<<<(110592 + 255) / 256, 256, 0, stream>>>(Win,  win16, 110592);
  cvt_kernel<<<(36864  + 255) / 256, 256, 0, stream>>>(Wout, wout16, 36864);
  cvt_kernel<<<(147456 + 255) / 256, 256, 0, stream>>>(Wf1,  wf1_16, 147456);
  cvt_kernel<<<(147456 + 255) / 256, 256, 0, stream>>>(Wf2,  wf2_16, 147456);

  // LN3 + window partition -> h f16
  ln_kernel<<<NT / 8, 256, 0, stream>>>(x, n3g, n3b, r1, 1);

  // QKV projection
  GemmOut io0 = {}; io0.q = qb; io0.k = kbf; io0.vt = vtb;
  gemm_kernel<0><<<dim3(NT / 64, 576 / 64), 128, 0, stream>>>(r1, win16, bin, 576, 192, io0);

  // block-causal flash attention -> attn_out (reuses r1)
  attn_kernel<<<576 * 8, 128, 0, stream>>>(qb, kbf, vtb, r1);

  // out-proj + window reverse + residual -> y f32
  GemmOut io1 = {}; io1.x = x; io1.y = y;
  gemm_kernel<1><<<dim3(NT / 64, 192 / 64), 128, 0, stream>>>(r1, wout16, bout, 192, 192, io1);

  // LN4 -> h2 f16 (flat token order, reuses r1)
  ln_kernel<<<NT / 8, 256, 0, stream>>>(y, n4g, n4b, r1, 0);

  // fc1 + GELU -> hidden f16 (reuses r2)
  GemmOut io2 = {}; io2.oh = r2;
  gemm_kernel<2><<<dim3(NT / 64, 768 / 64), 128, 0, stream>>>(r1, wf1_16, bf1, 768, 192, io2);

  // fc2 + residual -> d_out f32
  GemmOut io3 = {}; io3.resid = y; io3.of = out;
  gemm_kernel<3><<<dim3(NT / 64, 192 / 64), 128, 0, stream>>>(r2, wf2_16, bf2, 192, 768, io3);
}
